// PerceiverIO_26474178412628
// MI455X (gfx1250) — matrix operand order repaired and checked by reading, not compiled, not run on hardware
//
// PerceiverIO forward for MI455X (gfx1250), wave32 + WMMA bf16.
// ~1.3 TFLOP of matrix work vs ~0.3 ms HBM floor at 23.3 TB/s -> compute-bound.
// All GEMMs route through v_wmma_f32_16x16x32_bf16 (bf16 keeps fp32 exponent
// range, f32 accumulate). GEMM v2: vectorized b128 staging, b128 LDS fragment
// loads (padded/transposed LDS layouts), 2 accumulators per wave.

#include <hip/hip_runtime.h>
#include <hip/hip_bf16.h>
#include <math.h>

#define D_MODEL 512
#define DEPTH_N 20

typedef __attribute__((ext_vector_type(16))) __bf16 v16bf;
typedef __attribute__((ext_vector_type(16))) unsigned short v16us;
typedef __attribute__((ext_vector_type(8)))  float v8f;
typedef __attribute__((ext_vector_type(4)))  float v4f;
typedef __attribute__((ext_vector_type(4)))  unsigned int v4u;

__device__ __forceinline__ unsigned short f2bf(float f) {
  unsigned int u = __float_as_uint(f);
  unsigned int r = u + 0x7FFFu + ((u >> 16) & 1u);   // round-to-nearest-even
  return (unsigned short)(r >> 16);
}
__device__ __forceinline__ unsigned int pk2(float a, float b) {
  return (unsigned int)f2bf(a) | ((unsigned int)f2bf(b) << 16);
}

// ---------------------------------------------------------------------------
// Generic batched GEMM:  C[z] = alpha * A[z] x B[z]  (+bias) (+res)
// A: M x K (lda), B: K x N (ldb) or transB -> B[n*ldb+k], C: M x N (ldc).
// Batch z = zb*H + zh with independent b/h strides.
// 64x64 macro tile, 8 waves; each wave -> 16x32 output = 2 x wmma per K step.
// LDS: A row-major [64][40] (80B row stride, 16B-aligned 8-ushort runs),
//      B transposed [n][k] = [64][40] so each of a lane's two 8-K runs is contiguous.
// ---------------------------------------------------------------------------
#define TBM 64
#define TBN 64
#define TBK 32
#define LDP 40   // padded ushort row stride (80 B) keeps b128 LDS alignment

__global__ __launch_bounds__(256) void gemm_bf16_wmma(
    const float* __restrict__ A, const float* __restrict__ B,
    const float* __restrict__ bias, const float* __restrict__ res,
    float* __restrict__ C,
    int M, int N, int K,
    long long lda, long long ldb, long long ldc,
    int transB, float alpha, int H,
    long long sAb, long long sAh, long long sBb, long long sBh,
    long long sCb, long long sCh)
{
  __shared__ unsigned short As[TBM][LDP];   // [m][k]
  __shared__ unsigned short Bt[TBN][LDP];   // [n][k]  (transposed in LDS)

  const int tid = threadIdx.x;
  const int z  = blockIdx.z;
  const int zb = z / H, zh = z % H;
  A += (long long)zb * sAb + (long long)zh * sAh;
  B += (long long)zb * sBb + (long long)zh * sBh;
  const long long cOfs = (long long)zb * sCb + (long long)zh * sCh;

  const int m0 = blockIdx.y * TBM;
  const int n0 = blockIdx.x * TBN;

  const int wave = tid >> 5;
  const int lane = tid & 31;
  const int wm = wave >> 1;   // 0..3 : 16-row subtile
  const int wn = wave & 1;    // 0..1 : 32-col strip (two 16-col subtiles)

  const bool inb = (m0 + TBM <= M) && (n0 + TBN <= N) &&
                   ((lda & 3) == 0) && ((ldb & 3) == 0);

  v8f acc0 = {}, acc1 = {};

  for (int k0 = 0; k0 < K; k0 += TBK) {
    if (inb && (k0 + TBK <= K)) {
      // ---------- fast path: b128 global loads, packed LDS stores ----------
      {
        // A: thread -> row r, 8 consecutive k
        int r  = tid >> 2;
        int kk = (tid & 3) * 8;
        const float* ap = &A[(long long)(m0 + r) * lda + k0 + kk];
        v4f f0 = *(const v4f*)ap;
        v4f f1 = *(const v4f*)(ap + 4);
        v4u p; p.x = pk2(f0.x, f0.y); p.y = pk2(f0.z, f0.w);
               p.z = pk2(f1.x, f1.y); p.w = pk2(f1.z, f1.w);
        *(v4u*)&As[r][kk] = p;
      }
      if (transB) {
        // B^T: thread -> col n, 8 consecutive k (contiguous in memory)
        int nn = tid >> 2;
        int kk = (tid & 3) * 8;
        const float* bp = &B[(long long)(n0 + nn) * ldb + k0 + kk];
        v4f f0 = *(const v4f*)bp;
        v4f f1 = *(const v4f*)(bp + 4);
        v4u p; p.x = pk2(f0.x, f0.y); p.y = pk2(f0.z, f0.w);
               p.z = pk2(f1.x, f1.y); p.w = pk2(f1.z, f1.w);
        *(v4u*)&Bt[nn][kk] = p;
      } else {
        // B row-major: thread -> row k, 8 consecutive n; scatter into Bt
        int kk = tid >> 3;
        int nn = (tid & 7) * 8;
        const float* bp = &B[(long long)(k0 + kk) * ldb + n0 + nn];
        v4f f0 = *(const v4f*)bp;
        v4f f1 = *(const v4f*)(bp + 4);
        Bt[nn + 0][kk] = f2bf(f0.x); Bt[nn + 1][kk] = f2bf(f0.y);
        Bt[nn + 2][kk] = f2bf(f0.z); Bt[nn + 3][kk] = f2bf(f0.w);
        Bt[nn + 4][kk] = f2bf(f1.x); Bt[nn + 5][kk] = f2bf(f1.y);
        Bt[nn + 6][kk] = f2bf(f1.z); Bt[nn + 7][kk] = f2bf(f1.w);
      }
      if (k0 + TBK < K)  // speculative prefetch of next K tile
        __builtin_prefetch(&A[(long long)(m0 + (tid >> 2)) * lda + k0 + TBK], 0, 1);
    } else {
      // ---------- edge path: guarded scalar staging ----------
      for (int e = tid; e < TBM * TBK; e += 256) {
        int r = e / TBK, kk = e % TBK;
        int gr = m0 + r, gk = k0 + kk;
        float v = (gr < M && gk < K) ? A[(long long)gr * lda + gk] : 0.0f;
        As[r][kk] = f2bf(v);
      }
      for (int e = tid; e < TBK * TBN; e += 256) {
        int kk = e / TBN, nn = e % TBN;
        int gk = k0 + kk, gn = n0 + nn;
        float v = 0.0f;
        if (gk < K && gn < N)
          v = transB ? B[(long long)gn * ldb + gk] : B[(long long)gk * ldb + gn];
        Bt[nn][kk] = f2bf(v);
      }
    }
    __syncthreads();

    // --- fragments per ISA 7.12.2 (wave32) ---
    // A lane: row = lane&15; group0 K 0-7 & 16-23, group1 K 8-15 & 24-31
    // B lane: col = lane&15; group0 K 0-7 & 16-23, group1 K 8-15 & 24-31
    const int r   = lane & 15;
    const int kb  = (lane >> 4) * 8;
    const int kb2 = (lane >> 4) * 16;
    struct P32 { v4u a, b; } pa, pb0, pb1;
    pa.a  = *(const v4u*)&As[wm * 16 + r][kb];
    pa.b  = *(const v4u*)&As[wm * 16 + r][16 + kb];
    pb0.a = *(const v4u*)&Bt[wn * 32 + r][kb];
    pb0.b = *(const v4u*)&Bt[wn * 32 + r][16 + kb];
    pb1.a = *(const v4u*)&Bt[wn * 32 + 16 + r][kb];
    pb1.b = *(const v4u*)&Bt[wn * 32 + 16 + r][16 + kb];
    v16bf av = __builtin_bit_cast(v16bf, pa);
    acc0 = __builtin_amdgcn_wmma_f32_16x16x32_bf16(
        false, av, false, __builtin_bit_cast(v16bf, pb0), (short)0, acc0, false, false);
    acc1 = __builtin_amdgcn_wmma_f32_16x16x32_bf16(
        false, av, false, __builtin_bit_cast(v16bf, pb1), (short)0, acc1, false, false);
    __syncthreads();
  }

  // --- epilogue: C layout — VGPR r holds M=r (lanes 0-15) / M=r+8 (16-31) ---
  const int nl   = lane & 15;
  const int mofs = (lane >> 4) * 8;
#pragma unroll
  for (int rr = 0; rr < 8; ++rr) {
    int gm  = m0 + wm * 16 + mofs + rr;
    int gn0 = n0 + wn * 32 + nl;
    int gn1 = gn0 + 16;
    if (gm < M && gn0 < N) {
      float v = acc0[rr] * alpha;
      if (bias) v += bias[gn0];
      long long ci = cOfs + (long long)gm * ldc + gn0;
      if (res) v += res[ci];
      C[ci] = v;
    }
    if (gm < M && gn1 < N) {
      float v = acc1[rr] * alpha;
      if (bias) v += bias[gn1];
      long long ci = cOfs + (long long)gm * ldc + gn1;
      if (res) v += res[ci];
      C[ci] = v;
    }
  }
}

// ---------------------------------------------------------------------------
// LayerNorm over last dim (one row per block)
// ---------------------------------------------------------------------------
__global__ __launch_bounds__(256) void ln_kernel(
    const float* __restrict__ x, const float* __restrict__ g,
    const float* __restrict__ b, float* __restrict__ y, int D)
{
  __shared__ float sh[256], sh2[256];
  const long long row = blockIdx.x;
  const float* xr = x + row * (long long)D;
  float s = 0.f, s2 = 0.f;
  for (int i = threadIdx.x; i < D; i += 256) { float v = xr[i]; s += v; s2 += v * v; }
  sh[threadIdx.x] = s; sh2[threadIdx.x] = s2;
  __syncthreads();
  for (int st = 128; st > 0; st >>= 1) {
    if (threadIdx.x < st) { sh[threadIdx.x] += sh[threadIdx.x + st]; sh2[threadIdx.x] += sh2[threadIdx.x + st]; }
    __syncthreads();
  }
  float m   = sh[0] / D;
  float var = sh2[0] / D - m * m;
  float inv = rsqrtf(var + 1e-5f);
  for (int i = threadIdx.x; i < D; i += 256)
    y[row * (long long)D + i] = (xr[i] - m) * inv * g[i] + b[i];
}

// ---------------------------------------------------------------------------
// Row softmax (in place), one row per block
// ---------------------------------------------------------------------------
__global__ __launch_bounds__(256) void softmax_kernel(float* __restrict__ x, int len)
{
  __shared__ float sh[256];
  const long long row = blockIdx.x;
  float* xr = x + row * (long long)len;
  float mx = -3.4e38f;
  for (int i = threadIdx.x; i < len; i += 256) mx = fmaxf(mx, xr[i]);
  sh[threadIdx.x] = mx; __syncthreads();
  for (int st = 128; st > 0; st >>= 1) {
    if (threadIdx.x < st) sh[threadIdx.x] = fmaxf(sh[threadIdx.x], sh[threadIdx.x + st]);
    __syncthreads();
  }
  mx = sh[0];
  __syncthreads();
  float s = 0.f;
  for (int i = threadIdx.x; i < len; i += 256) { float e = expf(xr[i] - mx); xr[i] = e; s += e; }
  sh[threadIdx.x] = s; __syncthreads();
  for (int st = 128; st > 0; st >>= 1) {
    if (threadIdx.x < st) sh[threadIdx.x] += sh[threadIdx.x + st];
    __syncthreads();
  }
  float inv = 1.f / sh[0];
  for (int i = threadIdx.x; i < len; i += 256) xr[i] *= inv;
}

// ---------------------------------------------------------------------------
// Elementwise kernels
// ---------------------------------------------------------------------------
__global__ void bcast_lat(const float* __restrict__ lat, float* __restrict__ x, long long total)
{
  long long i = (long long)blockIdx.x * 256 + threadIdx.x;
  if (i < total) x[i] = lat[i % (256LL * 512)];
}

__global__ void geglu_kernel(const float* __restrict__ ff1, float* __restrict__ out, long long total)
{
  long long i = (long long)blockIdx.x * 256 + threadIdx.x;
  if (i >= total) return;
  long long r = i / 2048; int c = (int)(i % 2048);
  float a = ff1[r * 4096 + c];
  float g = ff1[r * 4096 + 2048 + c];
  float gl = 0.5f * g * (1.0f + erff(g * 0.70710678118654752f));  // exact gelu
  out[i] = a * gl;
}

__global__ void gru_kernel(const float* __restrict__ gi, const float* __restrict__ gh,
                           const float* __restrict__ h, float* __restrict__ x, long long total)
{
  long long i = (long long)blockIdx.x * 256 + threadIdx.x;
  if (i >= total) return;
  long long r = i / 512; int d = (int)(i % 512);
  const float* gir = gi + r * 1536;
  const float* ghr = gh + r * 1536;
  float rg = 1.f / (1.f + expf(-(gir[d] + ghr[d])));
  float zg = 1.f / (1.f + expf(-(gir[512 + d] + ghr[512 + d])));
  float nn = tanhf(gir[1024 + d] + rg * ghr[1024 + d]);
  float nh = (1.f - zg) * nn + zg * h[i];
  x[i] += nh;
}

__global__ void init_state(float* __restrict__ st)
{
  int i = blockIdx.x * 256 + threadIdx.x;
  if (i < 32) { st[i] = 1.f; st[32 + i] = 0.f; st[64 + i] = 0.f; }
  if (i < 32000) st[96 + i] = 0.f;
}

// ---------------------------------------------------------------------------
// Halting step: Threefry-2x32 (JAX) bernoulli + accumulators; writes outputs.
// d_out layout: p_s[20,32] | y_s[20,32,1000] | p_m[32] | y_m[32,1000]
// ---------------------------------------------------------------------------
__device__ __forceinline__ unsigned int rotl32(unsigned int x, int n) {
  return (x << n) | (x >> (32 - n));
}
__device__ void threefry2x32(unsigned int k0, unsigned int k1,
                             unsigned int& x0, unsigned int& x1)
{
  unsigned int ks[3] = { k0, k1, k0 ^ k1 ^ 0x1BD11BDAu };
  const int R[8] = { 13, 15, 26, 6, 17, 29, 16, 24 };
  x0 += ks[0]; x1 += ks[1];
  for (int i = 0; i < 5; ++i) {
    const int base = (i & 1) ? 4 : 0;
    for (int j = 0; j < 4; ++j) { x0 += x1; x1 = rotl32(x1, R[base + j]); x1 ^= x0; }
    x0 += ks[(i + 1) % 3];
    x1 += ks[(i + 2) % 3] + (unsigned int)(i + 1);
  }
}

__global__ __launch_bounds__(256) void halt_kernel(
    const float* __restrict__ logits, float* __restrict__ st,
    float* __restrict__ out, int n)
{
  const int b = blockIdx.x;
  __shared__ float s_halt;
  float* un = st; float* ha = st + 32; float* pm = st + 64; float* ym = st + 96;
  if (threadIdx.x == 0) {
    float l0  = logits[(long long)b * 1001];
    float lam = (n == DEPTH_N - 1) ? 1.0f : 1.0f / (1.0f + expf(-l0));
    float pn  = un[b] * lam;
    un[b] = un[b] * (1.0f - lam);
    // k = fold_in(key(42), n); bits = counter-mode threefry over iota(32)
    unsigned int f0 = 0u, f1 = (unsigned int)n;
    threefry2x32(0u, 42u, f0, f1);
    unsigned int c0 = (b < 16) ? (unsigned int)b : (unsigned int)(b - 16);
    unsigned int c1 = (b < 16) ? (unsigned int)(b + 16) : (unsigned int)b;
    threefry2x32(f0, f1, c0, c1);
    unsigned int bits = (b < 16) ? c0 : c1;
    float u    = __uint_as_float((bits >> 9) | 0x3f800000u) - 1.0f;
    float bern = (u < lam) ? 1.0f : 0.0f;
    float halt = bern * (1.0f - ha[b]);
    pm[b] = pm[b] * (1.0f - halt) + pn * halt;
    ha[b] = ha[b] + halt;
    out[(long long)n * 32 + b] = pn;     // p_s
    out[640640LL + b] = pm[b];           // p_m (last step wins)
    s_halt = halt;
  }
  __syncthreads();
  float halt = s_halt;
  for (int j = threadIdx.x; j < 1000; j += 256) {
    float yn = logits[(long long)b * 1001 + 1 + j];
    float v  = ym[(long long)b * 1000 + j] * (1.0f - halt) + yn * halt;
    ym[(long long)b * 1000 + j] = v;
    out[640LL + ((long long)n * 32 + b) * 1000 + j] = yn;   // y_s
    out[640672LL + (long long)b * 1000 + j] = v;            // y_m
  }
}

// ---------------------------------------------------------------------------
// Host orchestration
// ---------------------------------------------------------------------------
enum {
  P_LATENTS = 0, P_LN_CA_G, P_LN_CA_B, P_WQ_CA, P_WKV_CA, P_WO_CA, P_BO_CA,
  P_LN_CFF_G, P_LN_CFF_B, P_W1_CFF, P_B1_CFF, P_W2_CFF, P_B2_CFF,
  P_LN_LAT_G, P_LN_LAT_B, P_LN_SA_G, P_LN_SA_B,
  P_WQ_SA, P_WKV_SA, P_WO_SA, P_BO_SA,
  P_WIH, P_WHH, P_BIH, P_BHH,
  P_LN_DEC_G, P_LN_DEC_B, P_WQ_DEC, P_WKV_DEC, P_WO_DEC, P_BO_DEC,
  P_LN_DFF_G, P_LN_DFF_B, P_W1_DFF, P_B1_DFF, P_W2_DFF, P_B2_DFF,
  P_WL, P_BL, P_COUNT
};

extern "C" void kernel_launch(void* const* d_in, const int* in_sizes, int n_in,
                              void* d_out, int out_size, void* d_ws, size_t ws_size,
                              hipStream_t stream)
{
  (void)in_sizes; (void)out_size; (void)ws_size;
  const float* context = (const float*)d_in[0];
  const float* oq      = (const float*)d_in[2];
  const float* P[P_COUNT];
  for (int i = 0; i < P_COUNT && (3 + i) < n_in; ++i) P[i] = (const float*)d_in[3 + i];
  float* out = (float*)d_out;
  float* ws  = (float*)d_ws;

  // -------- scratch arena (float offsets) --------
  long long o = 0;
  auto alloc = [&](long long n) { long long r = o; o += n; return r; };
  float* x    = ws + alloc(32LL * 256 * 512);
  float* xn   = ws + alloc(32LL * 256 * 512);
  float* xi   = ws + alloc(32LL * 256 * 512);
  float* qb   = ws + alloc(32LL * 256 * 512);
  float* att  = ws + alloc(32LL * 256 * 512);
  float* yb   = ws + alloc(32LL * 256 * 512);
  float* gi   = ws + alloc(32LL * 256 * 1536);
  float* gh   = ws + alloc(32LL * 256 * 1536);
  float* kv   = ws + alloc(32LL * 2048 * 512);        // max: CA kv
  float* big  = ws + alloc(32LL * 4 * 256 * 2048);    // sim (CA), overlaid:
  float* ff1  = big;                                  //   rows x 4096
  float* ffm  = big + 32LL * 256 * 4096;              //   rows x 2048
  float* oqn  = ws + alloc(32LL * 64 * 512);
  float* latd = ws + alloc(32LL * 64 * 512);
  float* lat2 = ws + alloc(32LL * 64 * 512);
  float* dq   = ws + alloc(32LL * 64 * 256);
  float* lg   = ws + alloc(32LL * 1001);
  float* st   = ws + alloc(32LL * 3 + 32LL * 1000);

  auto gemm = [&](const float* A, const float* B, const float* bias, const float* res,
                  float* C, int M, int N, int K,
                  long long lda, long long ldb, long long ldc, int transB, float alpha,
                  int H, int batch,
                  long long sAb, long long sAh, long long sBb, long long sBh,
                  long long sCb, long long sCh) {
    dim3 g((N + TBN - 1) / TBN, (M + TBM - 1) / TBM, batch);
    gemm_bf16_wmma<<<g, 256, 0, stream>>>(A, B, bias, res, C, M, N, K, lda, ldb, ldc,
                                          transB, alpha, H, sAb, sAh, sBb, sBh, sCb, sCh);
  };
  auto gemm1 = [&](const float* A, const float* B, const float* bias, const float* res,
                   float* C, int M, int N, int K,
                   long long lda, long long ldb, long long ldc, int transB) {
    gemm(A, B, bias, res, C, M, N, K, lda, ldb, ldc, transB, 1.0f, 1, 1, 0, 0, 0, 0, 0, 0);
  };
  auto ln = [&](const float* in, int pg, int pb, float* outp, long long rows) {
    ln_kernel<<<dim3((unsigned)rows), 256, 0, stream>>>(in, P[pg], P[pb], outp, 512);
  };
  auto ew = [&](long long total) { return dim3((unsigned)((total + 255) / 256)); };

  // ================= encoder (once) =================
  init_state<<<ew(32000), 256, 0, stream>>>(st);
  bcast_lat<<<ew(32LL * 256 * 512), 256, 0, stream>>>(P[P_LATENTS], x, 32LL * 256 * 512);

  // cross-attention: latents attend to context (H=4, dh=64, I=256)
  ln(x, P_LN_CA_G, P_LN_CA_B, xn, 32LL * 256);
  gemm1(xn, P[P_WQ_CA], nullptr, nullptr, qb, 8192, 256, 512, 512, 256, 256, 0);
  gemm1(context, P[P_WKV_CA], nullptr, nullptr, kv, 65536, 512, 512, 512, 512, 512, 0);
  gemm(qb, kv, nullptr, nullptr, big, 256, 2048, 64, 256, 512, 2048, 1, 0.125f,
       4, 128, 256LL * 256, 64, 2048LL * 512, 64, 4LL * 256 * 2048, 256LL * 2048);
  softmax_kernel<<<32 * 4 * 256, 256, 0, stream>>>(big, 2048);
  gemm(big, kv + 256, nullptr, nullptr, att, 256, 64, 2048, 2048, 512, 256, 0, 1.0f,
       4, 128, 4LL * 256 * 2048, 256LL * 2048, 2048LL * 512, 64, 256LL * 256, 64);
  gemm1(att, P[P_WO_CA], P[P_BO_CA], x, x, 8192, 512, 256, 256, 512, 512, 0);

  // encoder GEGLU FF
  ln(x, P_LN_CFF_G, P_LN_CFF_B, xi, 32LL * 256);
  gemm1(xi, P[P_W1_CFF], P[P_B1_CFF], nullptr, ff1, 8192, 4096, 512, 512, 4096, 4096, 0);
  geglu_kernel<<<ew(8192LL * 2048), 256, 0, stream>>>(ff1, ffm, 8192LL * 2048);
  gemm1(ffm, P[P_W2_CFF], P[P_B2_CFF], x, x, 8192, 512, 2048, 2048, 512, 512, 0);

  // decoder query norm + projection (loop-invariant)
  ln(oq, P_LN_DEC_G, P_LN_DEC_B, oqn, 32LL * 64);
  gemm1(oqn, P[P_WQ_DEC], nullptr, nullptr, dq, 2048, 256, 512, 512, 256, 256, 0);

  // ================= 20-step scan =================
  for (int n = 0; n < DEPTH_N; ++n) {
    ln(x,  P_LN_LAT_G, P_LN_LAT_B, xn, 32LL * 256);
    ln(xn, P_LN_SA_G,  P_LN_SA_B,  xi, 32LL * 256);

    // self-attention (H=8, dh=64, I=512)
    gemm1(xi, P[P_WQ_SA],  nullptr, nullptr, qb, 8192, 512,  512, 512, 512,  512,  0);
    gemm1(xi, P[P_WKV_SA], nullptr, nullptr, kv, 8192, 1024, 512, 512, 1024, 1024, 0);
    gemm(qb, kv, nullptr, nullptr, big, 256, 256, 64, 512, 1024, 256, 1, 0.125f,
         8, 256, 256LL * 512, 64, 256LL * 1024, 64, 8LL * 256 * 256, 256LL * 256);
    softmax_kernel<<<32 * 8 * 256, 256, 0, stream>>>(big, 256);
    gemm(big, kv + 512, nullptr, nullptr, att, 256, 64, 256, 256, 1024, 512, 0, 1.0f,
         8, 256, 8LL * 65536, 65536, 256LL * 1024, 64, 256LL * 512, 64);
    gemm1(att, P[P_WO_SA], P[P_BO_SA], xn, yb, 8192, 512, 512, 512, 512, 512, 0);

    // GRU:  gi = y @ Wih.T + bih ;  gh = xn @ Whh.T + bhh ;  x += gru(...)
    gemm1(yb, P[P_WIH], P[P_BIH], nullptr, gi, 8192, 1536, 512, 512, 512, 1536, 1);
    gemm1(xn, P[P_WHH], P[P_BHH], nullptr, gh, 8192, 1536, 512, 512, 512, 1536, 1);
    gru_kernel<<<ew(8192LL * 512), 256, 0, stream>>>(gi, gh, xn, x, 8192LL * 512);

    // decoder cross-attention (H=4, dh=64, I=256): oq_n attends to x
    gemm1(x, P[P_WKV_DEC], nullptr, nullptr, kv, 8192, 512, 512, 512, 512, 512, 0);
    gemm(dq, kv, nullptr, nullptr, big, 64, 256, 64, 256, 512, 256, 1, 0.125f,
         4, 128, 64LL * 256, 64, 256LL * 512, 64, 4LL * 64 * 256, 64LL * 256);
    softmax_kernel<<<32 * 4 * 64, 256, 0, stream>>>(big, 256);
    gemm(big, kv + 256, nullptr, nullptr, att, 64, 64, 256, 256, 512, 256, 0, 1.0f,
         4, 128, 4LL * 16384, 16384, 256LL * 512, 64, 64LL * 256, 64);
    gemm1(att, P[P_WO_DEC], P[P_BO_DEC], nullptr, latd, 2048, 512, 256, 256, 512, 512, 0);

    // decoder GEGLU FF (residual)
    ln(latd, P_LN_DFF_G, P_LN_DFF_B, lat2, 32LL * 64);
    gemm1(lat2, P[P_W1_DFF], P[P_B1_DFF], nullptr, ff1, 2048, 4096, 512, 512, 4096, 4096, 0);
    geglu_kernel<<<ew(2048LL * 2048), 256, 0, stream>>>(ff1, ffm, 2048LL * 2048);
    gemm1(ffm, P[P_W2_DFF], P[P_B2_DFF], latd, latd, 2048, 512, 2048, 2048, 512, 512, 0);

    // logits: (32 x 32768) @ (32768 x 1001) + bl
    gemm1(latd, P[P_WL], P[P_BL], nullptr, lg, 32, 1001, 32768, 32768, 1001, 1001, 0);

    // halting + output scatter
    halt_kernel<<<32, 256, 0, stream>>>(lg, st, out, n);
  }
}
